// MultiHeadAttention_16063177687633
// MI455X (gfx1250) — matrix operand order repaired and checked by reading, not compiled, not run on hardware
//
#include <hip/hip_runtime.h>
#include <hip/hip_bf16.h>

// MI455X / gfx1250: wave32, WMMA f32_16x16x32_f16 path + async global->LDS copies.
// Pipeline: [x @ w_attn + b] (f16 out, WMMA) -> flash attention (WMMA + online
// softmax, scores never hit memory) -> [y @ w_proj + b] (WMMA, f32 out).

typedef __attribute__((ext_vector_type(16))) _Float16 v16h;
typedef __attribute__((ext_vector_type(8)))  float    v8f;

union F16Frag {
  v16h     v;
  _Float16 h[16];
  uint4    q[2];
};

#define WMMA_F16(A, B, C) \
  __builtin_amdgcn_wmma_f32_16x16x32_f16(false, (A), false, (B), (short)0, (C), false, false)

// ---- CDNA5 async global->LDS copy (per-lane 16B; tracked by ASYNCcnt) ------
// VDST = 32-bit LDS byte offset (low 32 bits of a generic pointer into LDS),
// VADDR = 64-bit global address, "off" = no SADDR.
__device__ __forceinline__ void async_copy_b128(void* lds_dst, const void* gsrc) {
  unsigned       l = (unsigned)(uintptr_t)lds_dst;
  unsigned long long g = (unsigned long long)(uintptr_t)gsrc;
  asm volatile("global_load_async_to_lds_b128 %0, %1, off" : : "v"(l), "v"(g) : "memory");
}
__device__ __forceinline__ void async_wait_all() {
  asm volatile("s_wait_asynccnt 0x0" : : : "memory");
}

// ---------------------------------------------------------------------------
// GEMM: C[M,N] = A[M,K] @ B[K,N] + bias[N]
// A: f32 (VALU convert on stage) or f16 (async b128 copy); B: f32 -> f16.
// Block = 128 threads (4 waves). Block tile 64x64, wave tile 32x32 (4 WMMAs/K-step).
// LDS: As[m][k] 64x32, Bs[n][k] 64x32 (B staged transposed so WMMA B-fragments
// are two contiguous 16B ds loads per lane).
// ---------------------------------------------------------------------------
template <typename AT, typename OT>
__global__ __launch_bounds__(128)
void gemm_wmma_kernel(const AT* __restrict__ A, const float* __restrict__ B,
                      const float* __restrict__ bias, OT* __restrict__ C,
                      int M, int N, int K) {
  __shared__ __align__(16) _Float16 As[64][32];   // [m][k]
  __shared__ __align__(16) _Float16 Bs[64][32];   // [n][k]

  const int t    = threadIdx.x;
  const int lane = t & 31;
  const int wid  = t >> 5;          // 0..3
  const int wm   = wid >> 1;        // 0..1
  const int wn   = wid & 1;         // 0..1
  const int lm   = lane & 15;
  const int hi   = (lane >> 4) & 1;

  const int n0 = blockIdx.x * 64;
  const int m0 = blockIdx.y * 64;

  v8f c00 = {}, c01 = {}, c10 = {}, c11 = {};

  const int arow = t >> 1;          // 0..63
  const int akb  = (t & 1) * 16;    // 0 / 16
  const int bkr  = t >> 2;          // 0..31
  const int bnb  = (t & 3) * 16;    // 0,16,32,48

#pragma unroll 2
  for (int k0 = 0; k0 < K; k0 += 32) {
    // ---- stage A tile ----
    const AT* ap = A + (size_t)(m0 + arow) * K + (k0 + akb);
    if constexpr (sizeof(AT) == 2) {
      // f16 source: straight 2x16B async global->LDS copies
      async_copy_b128(&As[arow][akb],     ap);
      async_copy_b128(&As[arow][akb + 8], ap + 8);
    } else {
#pragma unroll
      for (int i = 0; i < 16; ++i) As[arow][akb + i] = (_Float16)ap[i];
    }
    // ---- stage B tile transposed -> Bs[n][k] (f32 -> f16 convert) ----
    const float* bp = B + (size_t)(k0 + bkr) * N + (n0 + bnb);
#pragma unroll
    for (int i = 0; i < 16; ++i) Bs[bnb + i][bkr] = (_Float16)bp[i];

    if (k0 + 32 < K) {  // hint next tiles toward the caches (global_prefetch_b8)
      __builtin_prefetch(ap + 32, 0, 0);
      __builtin_prefetch(bp + (size_t)32 * N, 0, 0);
    }
    if constexpr (sizeof(AT) == 2) async_wait_all();
    __syncthreads();

    // ---- fragments from LDS (two 16B ds loads each, contiguous k = 16*hi + s in A and B) ----
    F16Frag a0, a1, b0, b1;
    const _Float16* ar0 = &As[wm * 32 + lm][0];
    a0.q[0] = *(const uint4*)(ar0 + hi * 16);
    a0.q[1] = *(const uint4*)(ar0 + hi * 16 + 8);
    const _Float16* ar1 = &As[wm * 32 + 16 + lm][0];
    a1.q[0] = *(const uint4*)(ar1 + hi * 16);
    a1.q[1] = *(const uint4*)(ar1 + hi * 16 + 8);
    const _Float16* br0 = &Bs[wn * 32 + lm][hi * 16];
    b0.q[0] = *(const uint4*)(br0);
    b0.q[1] = *(const uint4*)(br0 + 8);
    const _Float16* br1 = &Bs[wn * 32 + 16 + lm][hi * 16];
    b1.q[0] = *(const uint4*)(br1);
    b1.q[1] = *(const uint4*)(br1 + 8);

    c00 = WMMA_F16(a0.v, b0.v, c00);
    c01 = WMMA_F16(a0.v, b1.v, c01);
    c10 = WMMA_F16(a1.v, b0.v, c10);
    c11 = WMMA_F16(a1.v, b1.v, c11);
    __syncthreads();
  }

  // ---- epilogue: bias + store (C/D layout: row m = r + 8*hi, col n = lm) ----
#pragma unroll
  for (int sm = 0; sm < 2; ++sm) {
#pragma unroll
    for (int sn = 0; sn < 2; ++sn) {
      v8f acc = (sm == 0) ? (sn == 0 ? c00 : c01) : (sn == 0 ? c10 : c11);
      const int n  = n0 + wn * 32 + sn * 16 + lm;
      const float bv = bias[n];
#pragma unroll
      for (int r = 0; r < 8; ++r) {
        const int m = m0 + wm * 32 + sm * 16 + r + hi * 8;
        C[(size_t)m * N + n] = (OT)(acc[r] + bv);
      }
    }
  }
}

// ---------------------------------------------------------------------------
// Flash attention over qkv (f16, layout [B*S, 3*1024] = [q|k|v]).
// Grid: (S/64, NH, B). Block = 128 threads = 4 waves; wave owns 16 query rows.
// K tiles staged with async global->LDS b128 copies; V staged transposed on
// the VALU. Scores stay in WMMA accumulators; online softmax via 16-lane
// shuffles; P transposed C->A layout through a per-wave LDS patch (same-wave
// DS ordering, no barrier needed). y written f16 [B*S, 1024].
// ---------------------------------------------------------------------------
__global__ __launch_bounds__(128)
void attn_wmma_kernel(const _Float16* __restrict__ qkv, _Float16* __restrict__ y) {
  constexpr int S = 2048, D3 = 3072, DM = 1024, HD = 64;
  __shared__ __align__(16) _Float16 Ks[32][64];     // [key][hd]
  __shared__ __align__(16) _Float16 Vt[64][32];     // [hd][key] (transposed)
  __shared__ __align__(16) _Float16 Pb[4][16][32];  // per-wave P transpose patch

  const int t    = threadIdx.x;
  const int lane = t & 31;
  const int wid  = t >> 5;
  const int lm   = lane & 15;
  const int hi   = (lane >> 4) & 1;
  const int b  = blockIdx.z;
  const int h  = blockIdx.y;
  const int qt = blockIdx.x;

  const size_t bbase = (size_t)b * S;

  // Q fragments for this wave's 16 rows (HD=64 -> two K=32 steps), direct from global
  const int qrow = qt * 64 + wid * 16 + lm;
  const _Float16* qp = qkv + (bbase + qrow) * D3 + h * HD;
  F16Frag q0, q1;
  q0.q[0] = *(const uint4*)(qp + hi * 16);
  q0.q[1] = *(const uint4*)(qp + hi * 16 + 8);
  q1.q[0] = *(const uint4*)(qp + 32 + hi * 16);
  q1.q[1] = *(const uint4*)(qp + 32 + hi * 16 + 8);

  float Mr[8], Lr[8];
#pragma unroll
  for (int r = 0; r < 8; ++r) { Mr[r] = -1.0e30f; Lr[r] = 0.0f; }
  v8f y0 = {}, y1 = {}, y2 = {}, y3 = {};

  const int skey  = t >> 2;        // 0..31
  const int spart = (t & 3) * 16;  // 0,16,32,48

  for (int kt = 0; kt < S; kt += 32) {
    // ---- stage K tile via async global->LDS (straight f16 copy) ----
    const _Float16* kp = qkv + (bbase + kt + skey) * D3 + DM + h * HD + spart;
    async_copy_b128(&Ks[skey][spart],     kp);
    async_copy_b128(&Ks[skey][spart + 8], kp + 8);
    // ---- stage V tile transposed: Vt[hd][key] ----
    const _Float16* vp = qkv + (bbase + kt + skey) * D3 + 2 * DM + h * HD + spart;
    F16Frag vt;
    vt.q[0] = *(const uint4*)(vp);
    vt.q[1] = *(const uint4*)(vp + 8);
#pragma unroll
    for (int i = 0; i < 16; ++i) Vt[spart + i][skey] = vt.h[i];
    async_wait_all();
    __syncthreads();

    // ---- scores: S[16q x 32k] via 4 WMMAs (K^T B-frags: Ks is [n=key][k=hd]) ----
    F16Frag bk00, bk01, bk10, bk11;
    const _Float16* kr0 = &Ks[lm][0];
    bk00.q[0] = *(const uint4*)(kr0 + hi * 16);
    bk00.q[1] = *(const uint4*)(kr0 + hi * 16 + 8);
    bk01.q[0] = *(const uint4*)(kr0 + 32 + hi * 16);
    bk01.q[1] = *(const uint4*)(kr0 + 32 + hi * 16 + 8);
    const _Float16* kr1 = &Ks[16 + lm][0];
    bk10.q[0] = *(const uint4*)(kr1 + hi * 16);
    bk10.q[1] = *(const uint4*)(kr1 + hi * 16 + 8);
    bk11.q[0] = *(const uint4*)(kr1 + 32 + hi * 16);
    bk11.q[1] = *(const uint4*)(kr1 + 32 + hi * 16 + 8);

    v8f s0 = {}, s1 = {};
    s0 = WMMA_F16(q0.v, bk00.v, s0);
    s0 = WMMA_F16(q1.v, bk01.v, s0);
    s1 = WMMA_F16(q0.v, bk10.v, s1);
    s1 = WMMA_F16(q1.v, bk11.v, s1);

    // ---- online softmax (rows live across the 16-lane half-groups) ----
    float tm[8];
#pragma unroll
    for (int r = 0; r < 8; ++r) {
      s0[r] *= 0.125f;  // 1/sqrt(64)
      s1[r] *= 0.125f;
      tm[r] = fmaxf(s0[r], s1[r]);
    }
#pragma unroll
    for (int mask = 1; mask < 16; mask <<= 1)
#pragma unroll
      for (int r = 0; r < 8; ++r)
        tm[r] = fmaxf(tm[r], __shfl_xor(tm[r], mask, 32));

    float al[8], ts[8];
#pragma unroll
    for (int r = 0; r < 8; ++r) {
      const float mn = fmaxf(Mr[r], tm[r]);
      al[r] = __expf(Mr[r] - mn);
      Mr[r] = mn;
      s0[r] = __expf(s0[r] - mn);
      s1[r] = __expf(s1[r] - mn);
      ts[r] = s0[r] + s1[r];
    }
#pragma unroll
    for (int mask = 1; mask < 16; mask <<= 1)
#pragma unroll
      for (int r = 0; r < 8; ++r)
        ts[r] += __shfl_xor(ts[r], mask, 32);
#pragma unroll
    for (int r = 0; r < 8; ++r) {
      Lr[r] = Lr[r] * al[r] + ts[r];
      y0[r] *= al[r]; y1[r] *= al[r]; y2[r] *= al[r]; y3[r] *= al[r];
    }

    // ---- transpose P (C-layout) -> A-layout via per-wave LDS patch ----
#pragma unroll
    for (int r = 0; r < 8; ++r) {
      Pb[wid][r + hi * 8][lm]      = (_Float16)s0[r];
      Pb[wid][r + hi * 8][16 + lm] = (_Float16)s1[r];
    }
    F16Frag fp;
    const _Float16* pr = &Pb[wid][lm][0];
    fp.q[0] = *(const uint4*)(pr + hi * 16);
    fp.q[1] = *(const uint4*)(pr + hi * 16 + 8);

    // ---- y += P @ V : 4 WMMAs over hd subtiles (Vt is [n=hd][k=key]) ----
    F16Frag fv0, fv1, fv2, fv3;
    { const _Float16* vr = &Vt[0  + lm][hi * 16]; fv0.q[0] = *(const uint4*)(vr); fv0.q[1] = *(const uint4*)(vr + 8); }
    { const _Float16* vr = &Vt[16 + lm][hi * 16]; fv1.q[0] = *(const uint4*)(vr); fv1.q[1] = *(const uint4*)(vr + 8); }
    { const _Float16* vr = &Vt[32 + lm][hi * 16]; fv2.q[0] = *(const uint4*)(vr); fv2.q[1] = *(const uint4*)(vr + 8); }
    { const _Float16* vr = &Vt[48 + lm][hi * 16]; fv3.q[0] = *(const uint4*)(vr); fv3.q[1] = *(const uint4*)(vr + 8); }
    y0 = WMMA_F16(fp.v, fv0.v, y0);
    y1 = WMMA_F16(fp.v, fv1.v, y1);
    y2 = WMMA_F16(fp.v, fv2.v, y2);
    y3 = WMMA_F16(fp.v, fv3.v, y3);
    __syncthreads();
  }

  // ---- finalize: divide by row sums, store f16 y[b*S+row][h*64 + hd] ----
#pragma unroll
  for (int r = 0; r < 8; ++r) {
    const float inv = 1.0f / Lr[r];
    const int row = qt * 64 + wid * 16 + r + hi * 8;
    _Float16* yp = y + (bbase + row) * (size_t)DM + h * HD + lm;
    yp[0]  = (_Float16)(y0[r] * inv);
    yp[16] = (_Float16)(y1[r] * inv);
    yp[32] = (_Float16)(y2[r] * inv);
    yp[48] = (_Float16)(y3[r] * inv);
  }
}

// ---------------------------------------------------------------------------
extern "C" void kernel_launch(void* const* d_in, const int* in_sizes, int n_in,
                              void* d_out, int out_size, void* d_ws, size_t ws_size,
                              hipStream_t stream) {
  const float* x      = (const float*)d_in[0];  // [4,2048,1024]
  const float* w_attn = (const float*)d_in[1];  // [1024,3072]
  const float* b_attn = (const float*)d_in[2];  // [3072]
  const float* w_proj = (const float*)d_in[3];  // [1024,1024]
  const float* b_proj = (const float*)d_in[4];  // [1024]
  float* out = (float*)d_out;                   // [4,2048,1024] f32

  constexpr int M = 4 * 2048;  // 8192 rows
  // Workspace: qkv f16 (48 MB) + y f16 (16 MB)
  _Float16* qkv  = (_Float16*)d_ws;
  _Float16* ybuf = (_Float16*)((char*)d_ws + (size_t)M * 3072 * sizeof(_Float16));

  dim3 blk(128);
  // 1) qkv = x @ w_attn + b_attn  (f32 in, f16 out)
  gemm_wmma_kernel<float, _Float16>
      <<<dim3(3072 / 64, M / 64), blk, 0, stream>>>(x, w_attn, b_attn, qkv, M, 3072, 1024);
  // 2) flash attention -> ybuf (f16)
  attn_wmma_kernel<<<dim3(2048 / 64, 16, 4), blk, 0, stream>>>(qkv, ybuf);
  // 3) out = ybuf @ w_proj + b_proj  (f16 in, f32 out)
  gemm_wmma_kernel<_Float16, float>
      <<<dim3(1024 / 64, M / 64), blk, 0, stream>>>(ybuf, w_proj, b_proj, out, M, 1024, 1024);
}
